// MAttn2dForward_7662221656230
// MI455X (gfx1250) — compile-verified, matrix operand order checked by reading, not run on hardware
//
#include <hip/hip_runtime.h>
#include <cstdint>

// ---------------------------------------------------------------------------
// MAttn2d forward for MI455X (gfx1250): bf16 WMMA GEMMs + fp32 softmax.
// B=16, C=512, H=W=64 -> N=4096.
// ---------------------------------------------------------------------------

typedef __bf16 bf16_t;
typedef __attribute__((ext_vector_type(16))) __bf16 v16bf;
typedef __attribute__((ext_vector_type(8)))  __bf16 v8bf;
typedef __attribute__((ext_vector_type(8)))  float  v8f;

union Frag { v16bf v; v8bf h[2]; };

#define LP 40  // LDS row pitch in bf16 elements (32 data + 8 pad -> conflict-free b128 reads)

// gfx1250 async global->LDS path (ASYNCcnt-tracked, no VGPR round trip)
#if defined(__gfx1250__) && __has_builtin(__builtin_amdgcn_global_load_async_to_lds_b128)
#define ASYNC_LDS 1
#else
#define ASYNC_LDS 0
#endif

__device__ __forceinline__ void wait_async_lds() {
#if ASYNC_LDS
#if __has_builtin(__builtin_amdgcn_s_wait_asynccnt)
  __builtin_amdgcn_s_wait_asynccnt(0);
#else
  asm volatile("s_wait_asynccnt 0x0" ::: "memory");
#endif
#endif
}

#if ASYNC_LDS
typedef int v4i_ __attribute__((__vector_size__(16)));
typedef __attribute__((address_space(1))) v4i_* g_v4i_ptr;   // global
typedef __attribute__((address_space(3))) v4i_* l_v4i_ptr;   // LDS

__device__ __forceinline__ void async_copy_b128(const bf16_t* g, bf16_t* l) {
  // generic LDS pointer low 32 bits == LDS byte offset; global generic == AS1
  __builtin_amdgcn_global_load_async_to_lds_b128(
      (g_v4i_ptr)(unsigned long long)g,
      (l_v4i_ptr)(unsigned int)(unsigned long long)l,
      0, 0);
}
#endif

// ---------------------------------------------------------------------------
// Tiled bf16 WMMA GEMM:  C[M,N] = scale * (A[M,K] x B) (+ bias[m])
//   NT=0: B is row-major K x N (staged transposed into LDS)
//   NT=1: B is row-major N x K (C = A * B^T), staged directly
// Block 256 threads (8 waves), block tile 128x128, wave tile 64x32,
// 8 x v_wmma_f32_16x16x32_bf16 per K-step of 32. Dims must be multiples
// of (M:128, N:128, K:32) -- all call sites satisfy this.
// ---------------------------------------------------------------------------
template<int NT, int OUTBF16, int HASBIAS>
__global__ __launch_bounds__(256)
void gemm_wmma(const bf16_t* __restrict__ A, const bf16_t* __restrict__ Bm,
               const float* __restrict__ bias, void* __restrict__ Cout,
               int M, int N, int K, int lda, int ldb, int ldc,
               long long sA, long long sB, long long sC, float scale)
{
  __shared__ bf16_t As[2][128 * LP];
  __shared__ bf16_t Bs[2][128 * LP];

  const int z = blockIdx.z;
  A  += (long long)z * sA;
  Bm += (long long)z * sB;

  const int tileM = blockIdx.y * 128;
  const int tileN = blockIdx.x * 128;
  const int tid   = threadIdx.x;
  const int lane  = tid & 31;
  const int wave  = tid >> 5;     // 0..7
  const int wm    = wave & 1;     // 2 waves along M
  const int wn    = wave >> 1;    // 4 waves along N
  const int mrow  = lane & 15;    // row/col within 16x16 fragment
  const int koff  = (lane >> 4) * 8;

  // A-tile staging: 128 rows x 32 k, two b128 chunks per thread
  auto stageA = [&](int buf, int k0) {
#if ASYNC_LDS
#pragma unroll
    for (int i = 0; i < 2; ++i) {
      int c = tid + i * 256;
      int r = c >> 2, kc = (c & 3) * 8;
      async_copy_b128(A + (size_t)(tileM + r) * lda + (k0 + kc),
                      &As[buf][r * LP + kc]);
    }
#else
    int c0 = tid, c1 = tid + 256;
    int r0 = c0 >> 2, kc0 = (c0 & 3) * 8;
    int r1 = c1 >> 2, kc1 = (c1 & 3) * 8;
    v8bf t0 = *(const v8bf*)(A + (size_t)(tileM + r0) * lda + (k0 + kc0));
    v8bf t1 = *(const v8bf*)(A + (size_t)(tileM + r1) * lda + (k0 + kc1));
    *(v8bf*)(&As[buf][r0 * LP + kc0]) = t0;
    *(v8bf*)(&As[buf][r1 * LP + kc1]) = t1;
#endif
  };

  auto stageB = [&](int buf, int k0) {
    if (NT) {
      // B row-major (N x K): identical staging to A, rows indexed by n
#if ASYNC_LDS
#pragma unroll
      for (int i = 0; i < 2; ++i) {
        int c = tid + i * 256;
        int r = c >> 2, kc = (c & 3) * 8;
        async_copy_b128(Bm + (size_t)(tileN + r) * ldb + (k0 + kc),
                        &Bs[buf][r * LP + kc]);
      }
#else
      int c0 = tid, c1 = tid + 256;
      int r0 = c0 >> 2, kc0 = (c0 & 3) * 8;
      int r1 = c1 >> 2, kc1 = (c1 & 3) * 8;
      v8bf t0 = *(const v8bf*)(Bm + (size_t)(tileN + r0) * ldb + (k0 + kc0));
      v8bf t1 = *(const v8bf*)(Bm + (size_t)(tileN + r1) * ldb + (k0 + kc1));
      *(v8bf*)(&Bs[buf][r0 * LP + kc0]) = t0;
      *(v8bf*)(&Bs[buf][r1 * LP + kc1]) = t1;
#endif
    } else {
      // B row-major (K x N): transpose into Bs[n][k].
      // Issue both global loads before the LDS scatter for MLP.
      int c0 = tid, c1 = tid + 256;
      int k0i = c0 >> 4, nc0 = c0 & 15;
      int k1i = c1 >> 4, nc1 = c1 & 15;
      v8bf t0 = *(const v8bf*)(Bm + (size_t)(k0 + k0i) * ldb + tileN + nc0 * 8);
      v8bf t1 = *(const v8bf*)(Bm + (size_t)(k0 + k1i) * ldb + tileN + nc1 * 8);
#pragma unroll
      for (int j = 0; j < 8; ++j)
        Bs[buf][(nc0 * 8 + j) * LP + k0i] = t0[j];
#pragma unroll
      for (int j = 0; j < 8; ++j)
        Bs[buf][(nc1 * 8 + j) * LP + k1i] = t1[j];
    }
  };

  v8f acc[4][2];
  const v8f vzero = {0.f, 0.f, 0.f, 0.f, 0.f, 0.f, 0.f, 0.f};
#pragma unroll
  for (int mi = 0; mi < 4; ++mi)
#pragma unroll
    for (int ni = 0; ni < 2; ++ni) acc[mi][ni] = vzero;

  const int nK = K >> 5;
  stageA(0, 0);
  stageB(0, 0);
  wait_async_lds();
  __syncthreads();

  for (int s = 0; s < nK; ++s) {
    const int cur = s & 1;
    if (s + 1 < nK) {
      stageA(cur ^ 1, (s + 1) << 5);
      stageB(cur ^ 1, (s + 1) << 5);
    }
    if (s + 2 < nK) {  // uniform branch: EXEC stays full for WMMA below
      int r = tid & 127;
      __builtin_prefetch(A + (size_t)(tileM + r) * lda + ((s + 2) << 5), 0, 1);
    }

    Frag a[4], b[2];
#pragma unroll
    for (int mi = 0; mi < 4; ++mi) {
      const bf16_t* p = &As[cur][(wm * 64 + mi * 16 + mrow) * LP + koff];
      a[mi].h[0] = *(const v8bf*)(p);
      a[mi].h[1] = *(const v8bf*)(p + 16);
    }
#pragma unroll
    for (int ni = 0; ni < 2; ++ni) {
      const bf16_t* p = &Bs[cur][(wn * 32 + ni * 16 + mrow) * LP + koff];
      b[ni].h[0] = *(const v8bf*)(p);
      b[ni].h[1] = *(const v8bf*)(p + 16);
    }
#pragma unroll
    for (int mi = 0; mi < 4; ++mi)
#pragma unroll
      for (int ni = 0; ni < 2; ++ni)
        acc[mi][ni] = __builtin_amdgcn_wmma_f32_16x16x32_bf16(
            false, a[mi].v, false, b[ni].v, (short)0, acc[mi][ni], false, false);

    wait_async_lds();   // staged (cur^1) tile resident before anyone passes barrier
    __syncthreads();
  }

  // Epilogue: C/D layout -> VGPR r holds M = (lane>>4)*8 + r, N = lane&15
#pragma unroll
  for (int mi = 0; mi < 4; ++mi) {
#pragma unroll
    for (int ni = 0; ni < 2; ++ni) {
      const int col = tileN + wn * 32 + ni * 16 + mrow;
#pragma unroll
      for (int r = 0; r < 8; ++r) {
        const int row = tileM + wm * 64 + mi * 16 + koff + r;
        float v = acc[mi][ni][r] * scale;
        if (HASBIAS) v += bias[row];
        const size_t off = (size_t)z * sC + (size_t)row * ldc + col;
        if (OUTBF16) ((bf16_t*)Cout)[off] = (bf16_t)v;
        else         ((float*)Cout)[off]  = v;
      }
    }
  }
}

// ---------------------------------------------------------------------------
// Prep kernels
// ---------------------------------------------------------------------------

// Wqkv (1536x512 fp32) -> W5 rows [0,1536) bf16
__global__ void k_w5_qkv(const float* __restrict__ Wqkv, bf16_t* __restrict__ W5) {
  int i = blockIdx.x * 256 + threadIdx.x;
  if (i < 1536 * 512) W5[i] = (bf16_t)Wqkv[i];
}

// Wcat (1024x1024 bf16): rows [0,512)=Wg; rows [512,1024) = [0 | Wr]
__global__ void k_wcat(const float* __restrict__ Wg, const float* __restrict__ Wr,
                       bf16_t* __restrict__ Wcat) {
  int i = blockIdx.x * 256 + threadIdx.x;
  if (i >= 1024 * 1024) return;
  int row = i >> 10, j = i & 1023;
  float v;
  if (row < 512)      v = Wg[row * 1024 + j];
  else if (j < 512)   v = 0.0f;
  else                v = Wr[(row - 512) * 512 + (j - 512)];
  Wcat[i] = (bf16_t)v;
}

// b5 rows [0,1536) = bqkv ; bcat = [bg ; br]
__global__ void k_bias_copy(const float* __restrict__ bqkv, const float* __restrict__ bg,
                            const float* __restrict__ br, float* __restrict__ b5,
                            float* __restrict__ bcat) {
  int i = blockIdx.x * 256 + threadIdx.x;
  if (i < 1536) { b5[i] = bqkv[i]; return; }
  int j = i - 1536;
  if (j < 1024) bcat[j] = (j < 512) ? bg[j] : br[j - 512];
}

// W5 rows [1536,2560) = Wm (1024x512) @ Wqkv_v (512x512)
__global__ void k_wmq(const float* __restrict__ Wm, const float* __restrict__ Wqkv,
                      bf16_t* __restrict__ W5) {
  int i = blockIdx.x * 256 + threadIdx.x;
  if (i >= 1024 * 512) return;
  int o = i >> 9, c2 = i & 511;
  float s = 0.0f;
  for (int c = 0; c < 512; ++c)
    s += Wm[o * 512 + c] * Wqkv[(size_t)(1024 + c) * 512 + c2];
  W5[(size_t)(1536 + o) * 512 + c2] = (bf16_t)s;
}

// b5 rows [1536,2560) = bm + Wm @ bqkv_v
__global__ void k_wmq_bias(const float* __restrict__ Wm, const float* __restrict__ bqkv,
                           const float* __restrict__ bm, float* __restrict__ b5) {
  int o = blockIdx.x * 256 + threadIdx.x;
  if (o >= 1024) return;
  float s = bm[o];
  for (int c = 0; c < 512; ++c) s += Wm[o * 512 + c] * bqkv[1024 + c];
  b5[1536 + o] = s;
}

// x fp32 -> bf16 into cat rows [512,1024) per batch
__global__ void k_convert_x(const float* __restrict__ x, bf16_t* __restrict__ cat) {
  const size_t total = (size_t)16 * 512 * 4096;
  for (size_t i = blockIdx.x * 256ull + threadIdx.x; i < total; i += gridDim.x * 256ull) {
    size_t b = i >> 21, r = i & 2097151ull;
    cat[b * 4194304ull + 2097152ull + r] = (bf16_t)x[i];
  }
}

// _k = relu(k * mk), in place over k rows of O5
__global__ void k_relu_k(bf16_t* __restrict__ O5) {
  const size_t total = (size_t)16 * 512 * 4096;
  for (size_t i = blockIdx.x * 256ull + threadIdx.x; i < total; i += gridDim.x * 256ull) {
    size_t b = i >> 21, r = i & 2097151ull;
    size_t base = b * 10485760ull;                 // 2560*4096
    size_t kidx  = base + 2097152ull + r;          // rows [512,1024)
    size_t mkidx = base + 8388608ull + r;          // rows [2048,2560)
    float kv = (float)O5[kidx], mv = (float)O5[mkidx];
    O5[kidx] = (bf16_t)fmaxf(kv * mv, 0.0f);
  }
}

// _q = softmax_n(-relu(q*mq)) in place over q rows of O5 (row width 4096)
__global__ __launch_bounds__(256)
void k_softmax_q(bf16_t* __restrict__ O5) {
  const int rid = blockIdx.x;          // b*512 + c
  const int b = rid >> 9, c = rid & 511;
  bf16_t* qp = O5 + (size_t)b * 10485760ull + (size_t)c * 4096;
  const bf16_t* mp = qp + 6291456ull;  // rows 1536+c
  const int t = threadIdx.x;
  __shared__ float red[256];

  float sv[16];
  float mx = -1e30f;
#pragma unroll
  for (int i = 0; i < 16; ++i) {
    int idx = t + (i << 8);
    float s = -fmaxf((float)qp[idx] * (float)mp[idx], 0.0f);
    sv[i] = s;
    mx = fmaxf(mx, s);
  }
  red[t] = mx; __syncthreads();
  for (int o = 128; o > 0; o >>= 1) { if (t < o) red[t] = fmaxf(red[t], red[t + o]); __syncthreads(); }
  mx = red[0]; __syncthreads();

  float sum = 0.0f;
#pragma unroll
  for (int i = 0; i < 16; ++i) { sv[i] = __expf(sv[i] - mx); sum += sv[i]; }
  red[t] = sum; __syncthreads();
  for (int o = 128; o > 0; o >>= 1) { if (t < o) red[t] += red[t + o]; __syncthreads(); }
  const float inv = 1.0f / red[0];
#pragma unroll
  for (int i = 0; i < 16; ++i) qp[t + (i << 8)] = (bf16_t)(sv[i] * inv);
}

// attn = softmax_d(scores) (row width 512); write fp32 -> d_out, bf16 -> ws
__global__ __launch_bounds__(256)
void k_softmax_attn(const float* __restrict__ scores, float* __restrict__ attn_out,
                    bf16_t* __restrict__ attn_bf) {
  const int rid = blockIdx.x;          // b*512 + c
  const size_t base = (size_t)rid * 512;
  const int t = threadIdx.x;
  __shared__ float red[256];

  float s0 = scores[base + t], s1 = scores[base + t + 256];
  red[t] = fmaxf(s0, s1); __syncthreads();
  for (int o = 128; o > 0; o >>= 1) { if (t < o) red[t] = fmaxf(red[t], red[t + o]); __syncthreads(); }
  const float mx = red[0]; __syncthreads();

  float e0 = __expf(s0 - mx), e1 = __expf(s1 - mx);
  red[t] = e0 + e1; __syncthreads();
  for (int o = 128; o > 0; o >>= 1) { if (t < o) red[t] += red[t + o]; __syncthreads(); }
  const float inv = 1.0f / red[0];

  float a0 = e0 * inv, a1 = e1 * inv;
  attn_out[base + t]       = a0;
  attn_out[base + t + 256] = a1;
  attn_bf[base + t]        = (bf16_t)a0;
  attn_bf[base + t + 256]  = (bf16_t)a1;
}

// out = g*r + (1-g)*y, g = sigmoid(gr[0:512]), r = gr[512:1024], y from cat
__global__ void k_final(const float* __restrict__ grf, const bf16_t* __restrict__ cat,
                        float* __restrict__ out) {
  const size_t total = (size_t)16 * 512 * 4096;
  for (size_t i = blockIdx.x * 256ull + threadIdx.x; i < total; i += gridDim.x * 256ull) {
    size_t b = i >> 21, r = i & 2097151ull;
    size_t gbase = b * 4194304ull;
    float gv = grf[gbase + r];
    float rv = grf[gbase + 2097152ull + r];
    float yv = (float)cat[gbase + r];
    float g  = 1.0f / (1.0f + __expf(-gv));
    out[i] = g * rv + (1.0f - g) * yv;
  }
}

// ---------------------------------------------------------------------------
// Host launcher
// ---------------------------------------------------------------------------
extern "C" void kernel_launch(void* const* d_in, const int* in_sizes, int n_in,
                              void* d_out, int out_size, void* d_ws, size_t ws_size,
                              hipStream_t stream) {
  const float* x    = (const float*)d_in[0];
  const float* Wqkv = (const float*)d_in[1];
  const float* bqkv = (const float*)d_in[2];
  const float* Wm   = (const float*)d_in[3];
  const float* bm   = (const float*)d_in[4];
  const float* Wg   = (const float*)d_in[5];
  const float* bg   = (const float*)d_in[6];
  const float* Wr   = (const float*)d_in[7];
  const float* br   = (const float*)d_in[8];

  // workspace layout (256B aligned)
  char* ws = (char*)d_ws;
  size_t off = 0;
  auto alloc = [&](size_t bytes) { size_t r = off; off = (off + bytes + 255) & ~(size_t)255; return r; };
  const size_t o_W5   = alloc((size_t)2560 * 512 * 2);
  const size_t o_b5   = alloc((size_t)2560 * 4);
  const size_t o_Wcat = alloc((size_t)1024 * 1024 * 2);
  const size_t o_bcat = alloc((size_t)1024 * 4);
  const size_t o_cat  = alloc((size_t)16 * 1024 * 4096 * 2);  // [y ; x] bf16
  const size_t o_O5   = alloc((size_t)16 * 2560 * 4096 * 2);  // qkv+mqk bf16; later overlaid by gr fp32
  const size_t o_sc   = alloc((size_t)16 * 512 * 512 * 4);    // scores fp32
  const size_t o_abf  = alloc((size_t)16 * 512 * 512 * 2);    // attn bf16

  bf16_t* W5   = (bf16_t*)(ws + o_W5);
  float*  b5   = (float*)(ws + o_b5);
  bf16_t* Wcat = (bf16_t*)(ws + o_Wcat);
  float*  bcat = (float*)(ws + o_bcat);
  bf16_t* cat  = (bf16_t*)(ws + o_cat);
  bf16_t* O5   = (bf16_t*)(ws + o_O5);
  float*  grf  = (float*)(ws + o_O5);   // overlay: O5 dead once y computed
  float*  sc   = (float*)(ws + o_sc);
  bf16_t* abf  = (bf16_t*)(ws + o_abf);

  float* out_main = (float*)d_out;                      // 16*512*4096 fp32
  float* out_attn = out_main + (size_t)16 * 512 * 4096; // 16*512*512 fp32

  // --- weight / input prep ---
  k_w5_qkv   <<<(1536 * 512 + 255) / 256, 256, 0, stream>>>(Wqkv, W5);
  k_wcat     <<<(1024 * 1024 + 255) / 256, 256, 0, stream>>>(Wg, Wr, Wcat);
  k_bias_copy<<<(2560 + 255) / 256, 256, 0, stream>>>(bqkv, bg, br, b5, bcat);
  k_wmq      <<<(1024 * 512 + 255) / 256, 256, 0, stream>>>(Wm, Wqkv, W5);
  k_wmq_bias <<<4, 256, 0, stream>>>(Wm, bqkv, bm, b5);
  k_convert_x<<<16384, 256, 0, stream>>>(x, cat);

  const long long sO5  = 2560LL * 4096;  // per-batch O5 stride
  const long long sCat = 1024LL * 4096;
  const long long sSc  = 512LL * 512;

  // --- GEMM1: O5 = W5 (2560x512) @ xf (512x4096) + b5, bf16 out ---
  gemm_wmma<0, 1, 1><<<dim3(32, 20, 16), 256, 0, stream>>>(
      W5, cat + 512 * 4096, b5, O5,
      2560, 4096, 512, 512, 4096, 4096, 0, sCat, sO5, 1.0f);

  // --- elementwise: _k = relu(k*mk); _q = softmax(-relu(q*mq)) ---
  k_relu_k   <<<16384, 256, 0, stream>>>(O5);
  k_softmax_q<<<8192, 256, 0, stream>>>(O5);

  // --- GEMM2 (NT): scores = _q @ _k^T / sqrt(C), fp32 out ---
  gemm_wmma<1, 0, 0><<<dim3(4, 4, 16), 256, 0, stream>>>(
      O5, O5 + 512 * 4096, nullptr, sc,
      512, 512, 4096, 4096, 4096, 512, sO5, sO5, sSc, 0.044194173824159216f);

  // --- attn softmax (fp32 -> d_out, bf16 -> ws) ---
  k_softmax_attn<<<8192, 256, 0, stream>>>(sc, out_attn, abf);

  // --- GEMM3: y = attn (512x512) @ v (512x4096), bf16 into cat rows [0,512) ---
  gemm_wmma<0, 1, 0><<<dim3(32, 4, 16), 256, 0, stream>>>(
      abf, O5 + (size_t)1024 * 4096, nullptr, cat,
      512, 4096, 512, 512, 4096, 4096, sSc, sO5, sCat, 1.0f);

  // --- GEMM4: gr = Wcat (1024x1024) @ cat (1024x4096) + bcat, fp32 out ---
  gemm_wmma<0, 0, 1><<<dim3(32, 8, 16), 256, 0, stream>>>(
      Wcat, cat, bcat, grf,
      1024, 4096, 1024, 1024, 4096, 4096, 0, sCat, sCat, 1.0f);

  // --- final gate combine -> d_out ---
  k_final<<<16384, 256, 0, stream>>>(grf, cat, out_main);
}
